// SSM_67714454389069
// MI455X (gfx1250) — matrix operand order repaired and checked by reading, not compiled, not run on hardware
//
#include <hip/hip_runtime.h>
#include <hip/hip_bf16.h>

typedef __attribute__((ext_vector_type(16))) _Float16 v16h;
typedef __attribute__((ext_vector_type(8)))  _Float16 v8h;
typedef __attribute__((ext_vector_type(8)))  float    v8f;

#define NVOX  8000      // 20^3 voxels
#define CIN   64
#define CMID  16
#define DIM   20
#define UWIN  7
#define PAD   3
#define QPD   26        // 20 + 2*3 (zero-padded q)
#define KTOT  432       // 27 offsets * 16 channels
#define KSTEPS 14       // K padded to 448 = 14 * 32

// ---------------------------------------------------------------------------
// init: zero padded-q buffer and instance-norm stats accumulators
// ---------------------------------------------------------------------------
__global__ void k_init(unsigned* __restrict__ qp_u32, float* __restrict__ stats) {
  int i = blockIdx.x * 256 + threadIdx.x;
  const int nqp = QPD * QPD * QPD * CMID / 2;   // halves packed as u32
  if (i < nqp) qp_u32[i] = 0u;
  if (i < 2 * CIN) stats[i] = 0.f;
}

// ---------------------------------------------------------------------------
// channel reduction 64->16 (1x1x1 conv) + L2 normalize over channels, f16 out
// ---------------------------------------------------------------------------
__global__ void k_reduce(const float* __restrict__ x, const float* __restrict__ w_red,
                         _Float16* __restrict__ qp) {
  int v = blockIdx.x * 256 + threadIdx.x;
  if (v >= NVOX) return;
  float xr[CIN];
  for (int ci = 0; ci < CIN; ++ci) xr[ci] = x[ci * NVOX + v];
  float q[CMID];
  float ss = 0.f;
  for (int c = 0; c < CMID; ++c) {
    float a = 0.f;
    for (int ci = 0; ci < CIN; ++ci) a += w_red[c * CIN + ci] * xr[ci];
    q[c] = a; ss += a * a;
  }
  float inv = 1.f / fmaxf(sqrtf(ss), 1e-12f);
  int z = v / 400, y = (v / 20) % 20, xx = v % 20;
  _Float16* dst = qp + (((z + PAD) * QPD + (y + PAD)) * QPD + (xx + PAD)) * CMID;
  for (int c = 0; c < CMID; ++c) dst[c] = (_Float16)(q[c] * inv);
}

// ---------------------------------------------------------------------------
// pre-pack conv weights into WMMA B-fragment layout:
//   wfrag[layer][kstep][lane][h] = B[K][N], K = kstep*32 + (h&7) + 16*(h>=8) + 8*(lane>=16),
//   N = lane&15.  K decomposes as (offset=K>>4 -> dz,dy,dx ; cin=K&15).
// Also fold conv bias + BatchNorm(eval) into per-channel scale/bias.
// ---------------------------------------------------------------------------
__global__ void k_pack(const float* __restrict__ conv_w, const float* __restrict__ conv_b,
                       const float* __restrict__ bn_g, const float* __restrict__ bn_b,
                       const float* __restrict__ bn_m, const float* __restrict__ bn_v,
                       _Float16* __restrict__ wfrag, float* __restrict__ sb) {
  int i = blockIdx.x * 256 + threadIdx.x;
  const int total = 3 * KSTEPS * 32 * 16;
  if (i < total) {
    int h     = i & 15;
    int lane  = (i >> 4) & 31;
    int kstep = (i >> 9) % KSTEPS;
    int layer = i / (KSTEPS * 512);
    int k = kstep * 32 + (h & 7) + ((h & 8) ? 16 : 0) + ((lane & 16) ? 8 : 0);
    _Float16 val = (_Float16)0.f;
    if (k < KTOT) {
      int off = k >> 4, c = k & 15;
      int dz = off / 9, dy = (off / 3) % 3, dx = off % 3;
      int n = lane & 15;
      val = (_Float16)conv_w[((((layer * CMID + n) * CMID + c) * 3 + dz) * 3 + dy) * 3 + dx];
    }
    wfrag[i] = val;
  }
  if (i < 3 * CMID) {
    int layer = i / CMID, c = i % CMID;
    float s  = bn_g[layer * CMID + c] * rsqrtf(bn_v[layer * CMID + c] + 1e-5f);
    float bb = (conv_b[layer * CMID + c] - bn_m[layer * CMID + c]) * s + bn_b[layer * CMID + c];
    sb[layer * 32 + c]      = s;
    sb[layer * 32 + 16 + c] = bb;
  }
}

// ---------------------------------------------------------------------------
// compile-time LDS offset (in halves) for logical K index k, input dim IN:
//   k -> spatial offset (dz,dy,dx) = k>>4 in 3x3x3, channel c = k&15
// ---------------------------------------------------------------------------
template<int IN>
__device__ __host__ constexpr int koff(int k) {
  int off = k >> 4, c = k & 15;
  int dz = off / 9, dy = (off / 3) % 3, dx = off % 3;
  return ((dz * IN + dy) * IN + dx) * CMID + c;
}

// ---------------------------------------------------------------------------
// one 16x16 WMMA output tile of conv-as-GEMM (IN^3 window -> OUT^3 positions):
//   A row m = output position, A col K = (dz,dy,dx,cin).
//   16-bit A layout: M = lane&15, element h -> K = h + 8*(h>=8) + 8*(lane>=16).
//   Since slot K values have (K&15) <= 7, the lane-half "+8" in K is exactly
//   +8 halves in LDS, folded into the base address.  Each fragment is two
//   contiguous 16B chunks -> 2x ds_load_b128 with immediate offsets.
// ---------------------------------------------------------------------------
template<int IN, int OUT>
__device__ __forceinline__ v8f sse_tile(const _Float16* in_lds, int mBase, int mValid,
                                        const _Float16* __restrict__ wlayer, int lane) {
  int m = mBase + (lane & 15);
  if (m >= mValid) m = mValid - 1;           // padded rows: read safe, discard later
  int z0 = m / (OUT * OUT), y0 = (m / OUT) % OUT, x0 = m % OUT;
  int rowbase = ((z0 * IN + y0) * IN + x0) * CMID + ((lane & 16) ? 8 : 0);
  const _Float16* bp = wlayer + lane * 16;
  v8f acc = {};
#pragma unroll
  for (int ks = 0; ks < KSTEPS; ++ks) {
    union { v16h v; v8h h[2]; } A;
    A.h[0] = *(const v8h*)(in_lds + rowbase + koff<IN>(ks * 32));       // ds_load_b128
    if (ks * 32 + 16 < KTOT) {
      A.h[1] = *(const v8h*)(in_lds + rowbase + koff<IN>(ks * 32 + 16)); // ds_load_b128
    } else {
      v8h zero = {};
      A.h[1] = zero;                          // K padding (ks==13 upper chunk)
    }
    v16h b = *(const v16h*)(bp + ks * 512);   // 32B/lane, L2-broadcast
    acc = __builtin_amdgcn_wmma_f32_16x16x32_f16(
        false, A.v, false, b, (short)0, acc, false, false);
  }
  return acc;
}

// C/D layout: N = lane&15, M = mBase + r + 8*(lane>=16). Apply BN+ReLU, f16 out.
__device__ __forceinline__ void sse_store(v8f acc, int mBase, int mValid, int lane,
                                          const float* __restrict__ sbl,
                                          _Float16* out_lds) {
  float scale = sbl[lane & 15];
  float bias  = sbl[16 + (lane & 15)];
  int mAdd = (lane & 16) ? 8 : 0;
#pragma unroll
  for (int r = 0; r < 8; ++r) {
    int M = mBase + r + mAdd;
    if (M < mValid) {
      float val = acc[r] * scale + bias;
      out_lds[M * CMID + (lane & 15)] = (_Float16)(val > 0.f ? val : 0.f);
    }
  }
}

// ---------------------------------------------------------------------------
// fused per-voxel kernel: self-correlation window -> LDS, then 3 WMMA conv
// layers 7^3 -> 5^3 -> 3^3 -> 1, BN+ReLU folded, result -> feat[v][16]
// one block (4 wave32s) per voxel
// ---------------------------------------------------------------------------
__global__ __launch_bounds__(128)
void k_sse(const _Float16* __restrict__ qp, const _Float16* __restrict__ wfrag,
           const float* __restrict__ sb, float* __restrict__ feat) {
  __shared__ _Float16 s_sim[UWIN * UWIN * UWIN * CMID];  // 343*16 f16
  __shared__ _Float16 s_b1[5 * 5 * 5 * CMID];            // layer0 out
  __shared__ _Float16 s_b2[27 * CMID];                   // layer1 out
  __shared__ float    s_ctr[CMID];

  int v = blockIdx.x;
  int z = v / 400, y = (v / 20) % 20, xx = v % 20;
  int tid = threadIdx.x;
  int lane = tid & 31, wave = tid >> 5;

  __builtin_prefetch(wfrag, 0, 1);                        // global_prefetch_b8
  __builtin_prefetch(wfrag + KSTEPS * 512, 0, 1);
  __builtin_prefetch(wfrag + 2 * KSTEPS * 512, 0, 1);

  if (tid < CMID)
    s_ctr[tid] = (float)qp[(((z + PAD) * QPD + (y + PAD)) * QPD + (xx + PAD)) * CMID + tid];
  __syncthreads();

  // stage sim(o,c) = relu(q[p+o-P] * q[p]) into LDS
  for (int idx = tid; idx < UWIN * UWIN * UWIN * CMID; idx += 128) {
    int c = idx & 15;
    int o = idx >> 4;
    int dz = o / 49, dy = (o / 7) % 7, dx = o % 7;
    float w = (float)qp[(((z + dz) * QPD + (y + dy)) * QPD + (xx + dx)) * CMID + c];
    float p = w * s_ctr[c];
    s_sim[idx] = (_Float16)(p > 0.f ? p : 0.f);
  }
  __syncthreads();

  const _Float16* w0 = wfrag;
  const _Float16* w1 = wfrag + KSTEPS * 32 * 16;
  const _Float16* w2 = wfrag + 2 * KSTEPS * 32 * 16;

  // layer 0: 125 output rows -> 8 M-tiles, 2 per wave (112 WMMA / block)
  {
    v8f a0 = sse_tile<7, 5>(s_sim, wave * 16,       125, w0, lane);
    v8f a1 = sse_tile<7, 5>(s_sim, (wave + 4) * 16, 125, w0, lane);
    sse_store(a0, wave * 16,       125, lane, sb, s_b1);
    sse_store(a1, (wave + 4) * 16, 125, lane, sb, s_b1);
  }
  __syncthreads();

  // layer 1: 27 output rows -> 2 M-tiles on waves 0,1
  if (wave < 2) {
    v8f a = sse_tile<5, 3>(s_b1, wave * 16, 27, w1, lane);
    sse_store(a, wave * 16, 27, lane, sb + 32, s_b2);
  }
  __syncthreads();

  // layer 2: 1 output row -> wave 0, write feat directly
  if (wave == 0) {
    v8f a = sse_tile<3, 1>(s_b2, 0, 1, w2, lane);
    if (lane < 16) {
      float scale = sb[64 + lane], bias = sb[64 + 16 + lane];
      float val = a[0] * scale + bias;
      feat[v * CMID + lane] = val > 0.f ? val : 0.f;
    }
  }
}

// ---------------------------------------------------------------------------
// 1x1x1 conv 16->64 + per-channel sum/sumsq (block-reduced, atomics)
// ---------------------------------------------------------------------------
__global__ void k_out(const float* __restrict__ feat, const float* __restrict__ w_out,
                      const float* __restrict__ b_out, float* __restrict__ y0,
                      float* __restrict__ stats) {
  int c = blockIdx.y;
  int v = blockIdx.x * 256 + threadIdx.x;
  float val = 0.f;
  if (v < NVOX) {
    float a = b_out[c];
    const float* w = w_out + c * CMID;
    const float* f = feat + v * CMID;
#pragma unroll
    for (int k = 0; k < CMID; ++k) a += w[k] * f[k];
    y0[c * NVOX + v] = a;
    val = a;
  }
  __shared__ float rs[256], rq[256];
  rs[threadIdx.x] = val; rq[threadIdx.x] = val * val;
  __syncthreads();
  for (int s = 128; s > 0; s >>= 1) {
    if (threadIdx.x < s) {
      rs[threadIdx.x] += rs[threadIdx.x + s];
      rq[threadIdx.x] += rq[threadIdx.x + s];
    }
    __syncthreads();
  }
  if (threadIdx.x == 0) {
    atomicAdd(&stats[c], rs[0]);
    atomicAdd(&stats[CIN + c], rq[0]);
  }
}

// instance-norm + LeakyReLU(0.1) + residual
__global__ void k_post(const float* __restrict__ y0, const float* __restrict__ stats,
                       const float* __restrict__ x, float* __restrict__ t1) {
  int i = blockIdx.x * 256 + threadIdx.x;
  if (i >= CIN * NVOX) return;
  int c = i / NVOX;
  float mean = stats[c] * (1.f / NVOX);
  float var  = stats[CIN + c] * (1.f / NVOX) - mean * mean;
  float yv = (y0[i] - mean) * rsqrtf(var + 1e-5f);
  yv = yv > 0.f ? yv : 0.1f * yv;
  t1[i] = x[i] + yv;
}

// 1x1x1 conv 64->64 (+ optional LeakyReLU when slope >= 0)
__global__ void k_ffn(const float* __restrict__ in, const float* __restrict__ w,
                      const float* __restrict__ b, float* __restrict__ out, float slope) {
  int v = blockIdx.x * 256 + threadIdx.x;
  int c = blockIdx.y;
  if (v >= NVOX) return;
  float a = b[c];
  const float* wr = w + c * CIN;
  for (int k = 0; k < CIN; ++k) a += wr[k] * in[k * NVOX + v];
  if (slope >= 0.f) a = a > 0.f ? a : slope * a;
  out[c * NVOX + v] = a;
}

// ---------------------------------------------------------------------------
extern "C" void kernel_launch(void* const* d_in, const int* in_sizes, int n_in,
                              void* d_out, int out_size, void* d_ws, size_t ws_size,
                              hipStream_t stream) {
  const float* x      = (const float*)d_in[0];
  const float* w_red  = (const float*)d_in[1];
  const float* conv_w = (const float*)d_in[2];
  const float* conv_b = (const float*)d_in[3];
  const float* bn_g   = (const float*)d_in[4];
  const float* bn_b   = (const float*)d_in[5];
  const float* bn_m   = (const float*)d_in[6];
  const float* bn_v   = (const float*)d_in[7];
  const float* w_out  = (const float*)d_in[8];
  const float* b_out  = (const float*)d_in[9];
  const float* w_f1   = (const float*)d_in[10];
  const float* b_f1   = (const float*)d_in[11];
  const float* w_f2   = (const float*)d_in[12];
  const float* b_f2   = (const float*)d_in[13];
  float* out = (float*)d_out;

  char* ws = (char*)d_ws;
  size_t off = 0;
  auto take = [&](size_t bytes) { size_t o = off; off = (off + bytes + 255) & ~(size_t)255; return o; };
  _Float16* qp    = (_Float16*)(ws + take((size_t)QPD * QPD * QPD * CMID * 2));
  _Float16* wfrag = (_Float16*)(ws + take((size_t)3 * KSTEPS * 32 * 16 * 2));
  float*    sb    = (float*)   (ws + take((size_t)3 * 32 * 4));
  float*    feat  = (float*)   (ws + take((size_t)NVOX * CMID * 4));
  float*    y0    = (float*)   (ws + take((size_t)CIN * NVOX * 4));
  float*    stats = (float*)   (ws + take((size_t)2 * CIN * 4));
  float*    t1    = (float*)   (ws + take((size_t)CIN * NVOX * 4));
  float*    t2    = (float*)   (ws + take((size_t)CIN * NVOX * 4));

  const int nqp_u32 = QPD * QPD * QPD * CMID / 2;
  k_init  <<<(nqp_u32 + 255) / 256, 256, 0, stream>>>((unsigned*)qp, stats);
  k_reduce<<<(NVOX + 255) / 256,    256, 0, stream>>>(x, w_red, qp);
  k_pack  <<<(3 * KSTEPS * 512 + 255) / 256, 256, 0, stream>>>(
      conv_w, conv_b, bn_g, bn_b, bn_m, bn_v, wfrag, sb);
  k_sse   <<<NVOX, 128, 0, stream>>>(qp, wfrag, sb, feat);
  k_out   <<<dim3((NVOX + 255) / 256, CIN), 256, 0, stream>>>(feat, w_out, b_out, y0, stats);
  k_post  <<<(CIN * NVOX + 255) / 256, 256, 0, stream>>>(y0, stats, x, t1);
  k_ffn   <<<dim3((NVOX + 255) / 256, CIN), 256, 0, stream>>>(t1, w_f1, b_f1, t2, 0.01f);
  k_ffn   <<<dim3((NVOX + 255) / 256, CIN), 256, 0, stream>>>(t2, w_f2, b_f2, out, -1.0f);
}
